// GraphAttentionLayer_18640158065208
// MI455X (gfx1250) — matrix operand order repaired and checked by reading, not compiled, not run on hardware
//
#include <hip/hip_runtime.h>
#include <hip/hip_bf16.h>
#include <math.h>

// Problem constants (match reference)
#define FIN   256
#define HEADS 4
#define FOUT  64
#define COLS  (HEADS * FOUT)   // 256 output columns = H*F_OUT
#define NEG_SLOPE 0.2f

typedef __attribute__((ext_vector_type(16))) __bf16 v16bf;
typedef __attribute__((ext_vector_type(8)))  __bf16 v8bf;
typedef __attribute__((ext_vector_type(8)))  float  v8f;

// ---------------------------------------------------------------------------
// Utility kernels
// ---------------------------------------------------------------------------
__global__ void fill_f32(float* __restrict__ p, float v, int n) {
    int i = blockIdx.x * blockDim.x + threadIdx.x;
    if (i < n) p[i] = v;
}

// x[fp32] -> split bf16 (hi + residual lo) for near-fp32-accurate bf16 WMMA
__global__ void cvt_x_split(const float* __restrict__ x,
                            __bf16* __restrict__ xh, __bf16* __restrict__ xl,
                            int n) {
    int i = blockIdx.x * blockDim.x + threadIdx.x;
    if (i < n) {
        float v  = x[i];
        __bf16 h = (__bf16)v;
        xh[i] = h;
        xl[i] = (__bf16)(v - (float)h);
    }
}

// W [H, FIN, FOUT] fp32 -> Wt [COLS, FIN] split bf16 where Wt[c][k] = W[c/64][k][c%64]
// (transposed so each WMMA B-fragment is a contiguous 32B load per lane)
__global__ void cvt_w_split(const float* __restrict__ W,
                            __bf16* __restrict__ wth, __bf16* __restrict__ wtl) {
    int i = blockIdx.x * blockDim.x + threadIdx.x;   // COLS*FIN = 65536 total
    int c = i >> 8, k = i & 255;
    int hd = c >> 6, o = c & 63;
    float v  = W[(hd * FIN + k) * FOUT + o];
    __bf16 h = (__bf16)v;
    wth[i] = h;
    wtl[i] = (__bf16)(v - (float)h);
}

// ---------------------------------------------------------------------------
// Projection GEMM: h[N, 256] = x[N, 256] * B[256, 256]  (B = per-head W)
// One 16x16 output tile per wave per col-tile; split-bf16 -> 3 WMMAs / k-step.
// ---------------------------------------------------------------------------
__global__ __launch_bounds__(256) void gat_proj_wmma(
    const __bf16* __restrict__ xh, const __bf16* __restrict__ xl,
    const __bf16* __restrict__ wth, const __bf16* __restrict__ wtl,
    float* __restrict__ h, int nrows)
{
    const int lane    = threadIdx.x & 31;
    const int wave    = threadIdx.x >> 5;      // 8 waves per block
    const int rt      = blockIdx.x;            // 16-row tile
    const int rlocal  = lane & 15;
    const int hihalf  = lane >> 4;             // 0: lanes 0-15, 1: lanes 16-31

    int arow = rt * 16 + rlocal;
    if (arow >= nrows) arow = nrows - 1;       // clamp loads; stores guarded below

    const int ct0 = wave;                      // this wave's two column tiles
    const int ct1 = wave + 8;
    const int c0  = ct0 * 16 + rlocal;         // B column (lane-striped)
    const int c1  = ct1 * 16 + rlocal;

    v8f acc0 = {};
    v8f acc1 = {};

    const __bf16* xrh = xh + (size_t)arow * FIN;
    const __bf16* xrl = xl + (size_t)arow * FIN;

#pragma unroll
    for (int k0 = 0; k0 < FIN; k0 += 32) {
        // ---- A fragment (16x32 bf16): lane holds row rlocal;
        //      lanes 0-15: K = k0+[0..15]
        //      lanes16-31: K = k0+16+[0..15]
        const int ka = k0 + hihalf * 16;
        v8bf a0h = *(const v8bf*)(xrh + ka);
        v8bf a1h = *(const v8bf*)(xrh + ka + 8);
        v8bf a0l = *(const v8bf*)(xrl + ka);
        v8bf a1l = *(const v8bf*)(xrl + ka + 8);
        v16bf ah, al;
#pragma unroll
        for (int i = 0; i < 8; ++i) {
            ah[i] = a0h[i]; ah[i + 8] = a1h[i];
            al[i] = a0l[i]; al[i + 8] = a1l[i];
        }

        // ---- B fragments (32x16 bf16): lane holds column; K-half by lane group.
        //      Wt is [COLS][FIN] row-major -> contiguous 16-element (32B) loads.
        const int kb = k0 + hihalf * 16;
        v16bf b0h = *(const v16bf*)(wth + (size_t)c0 * FIN + kb);
        v16bf b0l = *(const v16bf*)(wtl + (size_t)c0 * FIN + kb);
        v16bf b1h = *(const v16bf*)(wth + (size_t)c1 * FIN + kb);
        v16bf b1l = *(const v16bf*)(wtl + (size_t)c1 * FIN + kb);

        // Split-precision accumulate: Ah*Bh + Ah*Bl + Al*Bh
        acc0 = __builtin_amdgcn_wmma_f32_16x16x32_bf16(false, ah, false, b0h, (short)0, acc0, false, false);
        acc0 = __builtin_amdgcn_wmma_f32_16x16x32_bf16(false, ah, false, b0l, (short)0, acc0, false, false);
        acc0 = __builtin_amdgcn_wmma_f32_16x16x32_bf16(false, al, false, b0h, (short)0, acc0, false, false);

        acc1 = __builtin_amdgcn_wmma_f32_16x16x32_bf16(false, ah, false, b1h, (short)0, acc1, false, false);
        acc1 = __builtin_amdgcn_wmma_f32_16x16x32_bf16(false, ah, false, b1l, (short)0, acc1, false, false);
        acc1 = __builtin_amdgcn_wmma_f32_16x16x32_bf16(false, al, false, b1h, (short)0, acc1, false, false);
    }

    // ---- Store C/D: VGPR r -> row (hihalf*8 + r), col = lane&15 within tile
    const int col0 = ct0 * 16 + rlocal;
    const int col1 = ct1 * 16 + rlocal;
#pragma unroll
    for (int r = 0; r < 8; ++r) {
        int row = rt * 16 + hihalf * 8 + r;
        if (row < nrows) {
            h[(size_t)row * COLS + col0] = acc0[r];
            h[(size_t)row * COLS + col1] = acc1[r];
        }
    }
}

// ---------------------------------------------------------------------------
// Per-node attention logits: e_s[n,hd] = h[n,hd,:] . a_src[hd,:], same for a_dst
// One wave per node: lane = hd*8 + j, each lane dots 8 features, shfl-reduce.
// ---------------------------------------------------------------------------
__global__ __launch_bounds__(256) void attn_logits(
    const float* __restrict__ h, const float* __restrict__ a_src,
    const float* __restrict__ a_dst, float* __restrict__ es,
    float* __restrict__ ed, int n)
{
    int wave = threadIdx.x >> 5, lane = threadIdx.x & 31;
    int node = blockIdx.x * 8 + wave;
    if (node >= n) return;
    int hd = lane >> 3, j = lane & 7;

    const float4* hp = (const float4*)(h + (size_t)node * COLS + hd * FOUT + j * 8);
    const float4* sp = (const float4*)(a_src + hd * FOUT + j * 8);
    const float4* dp = (const float4*)(a_dst + hd * FOUT + j * 8);
    float4 h0 = hp[0], h1 = hp[1];
    float4 s0 = sp[0], s1 = sp[1];
    float4 d0 = dp[0], d1 = dp[1];

    float ps = h0.x * s0.x + h0.y * s0.y + h0.z * s0.z + h0.w * s0.w
             + h1.x * s1.x + h1.y * s1.y + h1.z * s1.z + h1.w * s1.w;
    float pd = h0.x * d0.x + h0.y * d0.y + h0.z * d0.z + h0.w * d0.w
             + h1.x * d1.x + h1.y * d1.y + h1.z * d1.z + h1.w * d1.w;

#pragma unroll
    for (int m = 1; m < 8; m <<= 1) {
        ps += __shfl_xor(ps, m, 32);
        pd += __shfl_xor(pd, m, 32);
    }
    if (j == 0) {
        es[node * HEADS + hd] = ps;
        ed[node * HEADS + hd] = pd;
    }
}

// ---------------------------------------------------------------------------
// Float atomic max via monotonic int mapping (init must be -inf)
// ---------------------------------------------------------------------------
__device__ __forceinline__ void atomicMaxF(float* addr, float v) {
    if (v >= 0.0f) atomicMax((int*)addr, __float_as_int(v));
    else           atomicMin((unsigned int*)addr, (unsigned int)__float_as_int(v));
}

__device__ __forceinline__ float lrelu(float s) {
    return s > 0.0f ? s : NEG_SLOPE * s;
}

// Edge pass 1: scores = leaky(e_s[src] + e_d[dst]); segment max into m[dst]
__global__ void edge_scores(const float* __restrict__ es, const float* __restrict__ ed,
                            const int* __restrict__ src, const int* __restrict__ dst,
                            float* __restrict__ sc, float* __restrict__ m, int E)
{
    int e = blockIdx.x * blockDim.x + threadIdx.x;
    if (e >= E) return;
    int s = src[e], d = dst[e];
    float4 a = ((const float4*)es)[s];
    float4 b = ((const float4*)ed)[d];
    float4 sco;
    sco.x = lrelu(a.x + b.x);
    sco.y = lrelu(a.y + b.y);
    sco.z = lrelu(a.z + b.z);
    sco.w = lrelu(a.w + b.w);
    ((float4*)sc)[e] = sco;
    float* mp = m + d * HEADS;
    atomicMaxF(mp + 0, sco.x);
    atomicMaxF(mp + 1, sco.y);
    atomicMaxF(mp + 2, sco.z);
    atomicMaxF(mp + 3, sco.w);
}

// Edge pass 2: ex = exp(score - m[dst]); segment sum into den[dst]
__global__ void edge_expsum(const int* __restrict__ dst,
                            float* __restrict__ sc, const float* __restrict__ m,
                            float* __restrict__ den, int E)
{
    int e = blockIdx.x * blockDim.x + threadIdx.x;
    if (e >= E) return;
    int d = dst[e];
    float4 sco = ((const float4*)sc)[e];
    float4 mm  = ((const float4*)m)[d];
    float4 ex;
    ex.x = __expf(sco.x - mm.x);
    ex.y = __expf(sco.y - mm.y);
    ex.z = __expf(sco.z - mm.z);
    ex.w = __expf(sco.w - mm.w);
    ((float4*)sc)[e] = ex;
    float* dp = den + d * HEADS;
    atomicAdd(dp + 0, ex.x);
    atomicAdd(dp + 1, ex.y);
    atomicAdd(dp + 2, ex.z);
    atomicAdd(dp + 3, ex.w);
}

// Aggregation: out[dst,hd,:] += (ex/den) * h[src,hd,:]
// 16 lanes per (edge, head), each lane does a float4 (h fits in L2 -> gathers hit L2)
__global__ __launch_bounds__(256) void aggregate(
    const float* __restrict__ h, const float* __restrict__ sc,
    const float* __restrict__ den, const int* __restrict__ src,
    const int* __restrict__ dst, float* __restrict__ out, int E)
{
    int gu = blockIdx.x * 16 + (threadIdx.x >> 4);   // (edge, head) unit
    int j  = threadIdx.x & 15;
    if (gu >= E * HEADS) return;
    int e = gu >> 2, hd = gu & 3;
    int s = src[e], d = dst[e];
    float w = sc[e * HEADS + hd] / fmaxf(den[d * HEADS + hd], 1e-9f);
    float4 hv = *(const float4*)(h + (size_t)s * COLS + hd * FOUT + j * 4);
    float* op = out + (size_t)d * COLS + hd * FOUT + j * 4;
    atomicAdd(op + 0, w * hv.x);
    atomicAdd(op + 1, w * hv.y);
    atomicAdd(op + 2, w * hv.z);
    atomicAdd(op + 3, w * hv.w);
}

__global__ void relu_inplace(float* __restrict__ o, int n) {
    int i = blockIdx.x * blockDim.x + threadIdx.x;
    if (i < n) o[i] = fmaxf(o[i], 0.0f);
}

// ---------------------------------------------------------------------------
// Launcher
// ---------------------------------------------------------------------------
extern "C" void kernel_launch(void* const* d_in, const int* in_sizes, int n_in,
                              void* d_out, int out_size, void* d_ws, size_t ws_size,
                              hipStream_t stream) {
    const float* x     = (const float*)d_in[0];
    const int*   esrc  = (const int*)d_in[1];
    const int*   edst  = (const int*)d_in[2];
    const float* W     = (const float*)d_in[3];
    const float* a_src = (const float*)d_in[4];
    const float* a_dst = (const float*)d_in[5];
    float* out = (float*)d_out;

    const int N = in_sizes[0] / FIN;
    const int E = in_sizes[1];

    // Workspace carve-out (256B aligned)
    char* ws = (char*)d_ws;
    size_t off = 0;
    auto carve = [&](size_t bytes) -> char* {
        char* p = ws + off;
        off += (bytes + 255) & ~(size_t)255;
        return p;
    };
    __bf16* xh  = (__bf16*)carve((size_t)N * FIN * 2);
    __bf16* xl  = (__bf16*)carve((size_t)N * FIN * 2);
    __bf16* wth = (__bf16*)carve((size_t)COLS * FIN * 2);
    __bf16* wtl = (__bf16*)carve((size_t)COLS * FIN * 2);
    float*  hbf = (float*)carve((size_t)N * COLS * 4);
    float*  es  = (float*)carve((size_t)N * HEADS * 4);
    float*  ed  = (float*)carve((size_t)N * HEADS * 4);
    float*  m   = (float*)carve((size_t)N * HEADS * 4);
    float*  den = (float*)carve((size_t)N * HEADS * 4);
    float*  sc  = (float*)carve((size_t)E * HEADS * 4);

    const int NH  = N * HEADS;
    const int NC  = N * COLS;
    const int WEL = COLS * FIN;

    // init: m = -inf, den = 0, out = 0 (out is the atomic accumulation target)
    fill_f32<<<(NH + 255) / 256, 256, 0, stream>>>(m, -INFINITY, NH);
    fill_f32<<<(NH + 255) / 256, 256, 0, stream>>>(den, 0.0f, NH);
    fill_f32<<<(NC + 255) / 256, 256, 0, stream>>>(out, 0.0f, NC);

    // convert inputs to split-bf16
    cvt_x_split<<<(N * FIN + 255) / 256, 256, 0, stream>>>(x, xh, xl, N * FIN);
    cvt_w_split<<<(WEL + 255) / 256, 256, 0, stream>>>(W, wth, wtl);

    // projection GEMM (WMMA)
    gat_proj_wmma<<<(N + 15) / 16, 256, 0, stream>>>(xh, xl, wth, wtl, hbf, N);

    // attention logits
    attn_logits<<<(N + 7) / 8, 256, 0, stream>>>(hbf, a_src, a_dst, es, ed, N);

    // edge softmax
    edge_scores<<<(E + 255) / 256, 256, 0, stream>>>(es, ed, esrc, edst, sc, m, E);
    edge_expsum<<<(E + 255) / 256, 256, 0, stream>>>(edst, sc, m, den, E);

    // weighted aggregation into out, then relu
    long long units = (long long)E * HEADS;
    long long aggThreads = units * 16;
    aggregate<<<(int)((aggThreads + 255) / 256), 256, 0, stream>>>(hbf, sc, den, esrc, edst, out, E);
    relu_inplace<<<(NC + 255) / 256, 256, 0, stream>>>(out, NC);
}
